// SCOFFCell_76115410419724
// MI455X (gfx1250) — matrix operand order repaired and checked by reading, not compiled, not run on hardware
//
#include <hip/hip_runtime.h>
#include <hip/hip_bf16.h>
#include <stdint.h>

typedef _Float16 v16h __attribute__((ext_vector_type(16)));
typedef float    v8f  __attribute__((ext_vector_type(8)));

#define BSZ 2048
#define KU  8
#define SHD 128
#define NROWS (BSZ*KU)        /* 16384 */
#define NTIL  (NROWS/16)      /* 1024 */

/* d_out offsets (floats): hx_out, cx_out, mask, block_mask, temp_att */
#define O_HX   0
#define O_CXO  (BSZ*KU*SHD)
#define O_MASK (2*BSZ*KU*SHD)
#define O_BM   (3*BSZ*KU*SHD)
#define O_TA   (O_BM + BSZ*KU)

__device__ __forceinline__ float sigf(float x){ return 1.0f/(1.0f+__expf(-x)); }

/* JAX threefry2x32 (20 rounds, key-schedule injection every 4) */
__device__ __forceinline__ void tf2x32(uint32_t k0,uint32_t k1,uint32_t x0,uint32_t x1,
                                       uint32_t &y0,uint32_t &y1){
  uint32_t ks[3] = {k0, k1, k0 ^ k1 ^ 0x1BD11BDAu};
  x0 += ks[0]; x1 += ks[1];
  const int R0[4] = {13,15,26,6};
  const int R1[4] = {17,29,16,24};
  #pragma unroll
  for (int i=0;i<5;++i){
    const int* R = (i & 1) ? R1 : R0;
    #pragma unroll
    for (int j=0;j<4;++j){ x0 += x1; x1 = (x1<<R[j]) | (x1>>(32-R[j])); x1 ^= x0; }
    x0 += ks[(i+1)%3];
    x1 += ks[(i+2)%3] + (uint32_t)(i+1);
  }
  y0 = x0; y1 = x1;
}

/* gumbel[j] for the (16384,4) noise tensor, key = fold_in(key(0), 12345) */
__device__ __forceinline__ float gumbel_at(uint32_t j){
  uint32_t fk0, fk1; tf2x32(0u,0u,0u,12345u,fk0,fk1);
  uint32_t a,b,bits;
  if (j < 32768u){ tf2x32(fk0,fk1, j, j+32768u, a,b); bits=a; }
  else           { tf2x32(fk0,fk1, j-32768u, j, a,b); bits=b; }
  float u = __uint_as_float(0x3f800000u | (bits>>9)) - 1.0f;
  const float tiny = 1.1754943508222875e-38f;
  u = fmaxf(u + tiny, tiny);
  return -logf(-logf(u));
}

/* A-fragment intra-chunk K offset for 16-bit A (contiguous: k = 16*hi + s), i = half-pair idx */
__device__ __forceinline__ int akoff(int i, int hi){
  return 2*i + hi*16;
}

/* ---------------- K0: weight swizzle into f16 B-fragment layouts ------------- */
__global__ __launch_bounds__(256) void k0_swizzle(
    const float* __restrict__ W_ih, const float* __restrict__ W_hh,
    const float* __restrict__ b_ih, const float* __restrict__ b_hh,
    const float* __restrict__ Wq_c, const float* __restrict__ Wk_c,
    const float* __restrict__ Wv_c, const float* __restrict__ Wf_c,
    const float* __restrict__ Wg_c,
    _Float16* __restrict__ Bg, _Float16* __restrict__ Wc, float* __restrict__ biasP){
  int tid = blockIdx.x*256 + threadIdx.x;
  if (tid < 32768){
    /* gate weights, permuted columns p = s*16 + m*4 + g, tiles [kt 0..7][nt 0..127] */
    int lane = tid & 31, nt = (tid>>5)&127, kt = tid>>12;
    int mg = lane & 15, m = mg>>2, g = mg&3, gcol = g*128 + nt;
    int khalf = (lane>>4)*16;
    v16h f;
    #pragma unroll
    for (int v=0; v<8; ++v){
      int k = kt*32 + khalf + 2*v;
      float a0 = (k < 128) ? W_ih[(m*512+gcol)*128 + k]   : W_hh[(m*512+gcol)*128 + k-128];
      float a1 = (k < 128) ? W_ih[(m*512+gcol)*128 + k+1] : W_hh[(m*512+gcol)*128 + k-127];
      f[2*v] = (_Float16)a0; f[2*v+1] = (_Float16)a1;
    }
    *(v16h*)(Bg + (size_t)tid*16) = f;
  } else if (tid < 32768 + 5120){
    /* 5 comm matrices 128x128: mats q,k,v,f,g; tiles [mat][kt 0..3][nt 0..7] */
    int t = tid - 32768;
    int lane = t & 31, nt = (t>>5)&7, kt = (t>>8)&3, mat = t>>10;
    const float* W = (mat==0)?Wq_c:(mat==1)?Wk_c:(mat==2)?Wv_c:(mat==3)?Wf_c:Wg_c;
    int col = nt*16 + (lane&15), khalf = (lane>>4)*16;
    v16h f;
    #pragma unroll
    for (int v=0; v<8; ++v){
      int k = kt*32 + khalf + 2*v;
      f[2*v]   = (_Float16)W[k*128+col];
      f[2*v+1] = (_Float16)W[(k+1)*128+col];
    }
    *(v16h*)(Wc + (size_t)t*16) = f;
  } else if (tid < 32768 + 5120 + 2048){
    int p = tid - (32768+5120);
    int s = p>>4, mg = p&15, m = mg>>2, g = mg&3, gcol = g*128+s;
    biasP[p] = b_ih[m*512+gcol] + b_hh[m*512+gcol];
  }
}

/* ---------------- K1: input attention + mask + A-fragment staging ------------ */
__global__ __launch_bounds__(128) void k1_inputattn(
    const float* __restrict__ inp, const float* __restrict__ hx,
    const float* __restrict__ Wq_i, const float* __restrict__ Wk_i,
    const float* __restrict__ Wv_i, const float* __restrict__ W_read,
    _Float16* __restrict__ Aswz, float* __restrict__ hreadG, float* __restrict__ out){
  __shared__ float hxL[1024], inpL[1024];
  __shared__ float qsL[8*64], kslL[9*64], vsL[9*128];
  __shared__ float iattL[8*9], xuseL[8*128], mu[8];
  const int b = blockIdx.x, tid = threadIdx.x;

  for (int j=0;j<8;++j){
    hxL [tid+j*128] = hx [(size_t)b*1024 + tid + j*128];
    inpL[tid+j*128] = inp[(size_t)b*1024 + tid + j*128];
  }
  __syncthreads();

  /* qs = q@Wq_i, ksl = kv@Wk_i (row 8 of kv is zero) */
  for (int r=0;r<4;++r){
    int o = tid + r*128, u = o>>6, d = o&63;
    float s1=0.f, s2=0.f;
    for (int i=0;i<128;++i){ s1 += hxL[u*128+i]*Wq_i[i*64+d]; s2 += inpL[u*128+i]*Wk_i[i*64+d]; }
    qsL[o]=s1; kslL[o]=s2;
  }
  if (tid < 64) kslL[512+tid] = 0.f;
  for (int r=0;r<8;++r){
    int o = tid*8 + r, u = o>>7, d = o&127;
    float s=0.f;
    for (int i=0;i<128;++i) s += inpL[u*128+i]*Wv_i[i*128+d];
    vsL[o]=s;
  }
  vsL[1024+tid] = 0.f;
  __syncthreads();

  if (tid < 72){
    int u = tid/9, nn = tid%9;
    float s=0.f;
    for (int d=0; d<64; ++d) s += qsL[u*64+d]*kslL[nn*64+d];
    iattL[tid] = s*0.125f;
  }
  __syncthreads();
  if (tid < 8){
    float mx=-1e30f;
    for (int nn=0;nn<9;++nn) mx = fmaxf(mx, iattL[tid*9+nn]);
    float e[9], sum=0.f;
    for (int nn=0;nn<9;++nn){ e[nn]=__expf(iattL[tid*9+nn]-mx); sum+=e[nn]; }
    float inv = 1.0f/sum;
    for (int nn=0;nn<9;++nn) iattL[tid*9+nn] = e[nn]*inv;
  }
  __syncthreads();
  if (tid == 0){
    float a[8]; bool drop[8];
    for (int u=0;u<8;++u){ a[u]=iattL[u*9]; drop[u]=false; }
    for (int r=0;r<3;++r){            /* top-3 by value, ties -> lowest index */
      int bi=0; float bv=-1e30f;
      for (int u=0;u<8;++u) if (!drop[u] && a[u] > bv){ bv=a[u]; bi=u; }
      drop[bi]=true;
    }
    for (int u=0;u<8;++u){ mu[u] = drop[u]?0.f:1.f; out[O_BM + (size_t)b*8 + u] = mu[u]; }
  }
  __syncthreads();
  for (int r=0;r<8;++r){ int j = tid + r*128; out[O_MASK + (size_t)b*1024 + j] = mu[j>>7]; }

  /* inp_use = iatt @ vs */
  for (int r=0;r<8;++r){
    int o = tid*8 + r, u = o>>7, d = o&127;
    float s=0.f;
    for (int nn=0;nn<9;++nn) s += iattL[u*9+nn]*vsL[nn*128+d];
    xuseL[o]=s;
  }
  /* h_read = hx @ W_read */
  {
    int u = tid>>4, d = tid&15;
    float s=0.f;
    for (int ss=0; ss<128; ++ss) s += hxL[u*128+ss]*W_read[ss*16+d];
    hreadG[((size_t)b*8+u)*16 + d] = s;
  }
  __syncthreads();

  /* stage [x|h] into f16 A-fragment layout: tile mt=b/2, chunks kt 0..7 */
  {
    int u = tid & 7, kt = (tid>>3)&7, hi = tid>>6;
    int mt = b>>1, mloc = (b&1)*8 + u, lane = mloc + hi*16;
    const float* src = (kt < 4) ? &xuseL[u*128 + kt*32] : &hxL[u*128 + (kt-4)*32];
    v16h f;
    #pragma unroll
    for (int i=0;i<8;++i){
      int c = akoff(i,hi);
      f[2*i] = (_Float16)src[c]; f[2*i+1] = (_Float16)src[c+1];
    }
    *(v16h*)(Aswz + ((size_t)(mt*8+kt)*32 + lane)*16) = f;
  }
}

/* ---------------- K2: gate GEMM + LSTM + selection + comm attention ---------- */
__global__ __launch_bounds__(256) void k2_core(
    const float* __restrict__ hx, const float* __restrict__ cx,
    const float* __restrict__ bf_c, const float* __restrict__ bg_c,
    const float* __restrict__ W_write,
    const _Float16* __restrict__ Aswz, const _Float16* __restrict__ Bg,
    const _Float16* __restrict__ Wc, const float* __restrict__ biasP,
    const float* __restrict__ hreadG, float* __restrict__ out){
  __shared__ _Float16 htL[16*4*128];           /* h_t (all templates), f16 */
  __shared__ float cxL[16*128];
  __shared__ float cbL[16*128];                /* selected c_t */
  __shared__ _Float16 qcL[16*128], kcL[16*128], vcL[16*128];
  __shared__ float oL[16*128];
  __shared__ float wkL[16*4*16];
  __shared__ float bmL[16];
  __shared__ int   selL[16];

  const int tid = threadIdx.x;
  const int w = tid >> 5, lane = tid & 31;
  const int mg = lane & 15, m = mg>>2, g = mg&3, hi = lane>>4;
  const int mt = blockIdx.x, n0 = mt*16;

  for (int r=0;r<8;++r){ int idx = tid + r*256; cxL[idx] = cx[(size_t)n0*128 + idx]; }
  if (tid < 16) bmL[tid] = out[O_BM + n0 + tid];
  __syncthreads();

  /* ---- gate GEMM: [x|h](16x256) @ Wcat(256x2048-permuted); wave w owns s in [16w,16w+16) */
  v8f acc[16];
  #pragma unroll
  for (int t=0;t<16;++t){
    #pragma unroll
    for (int i=0;i<8;++i) acc[t][i]=0.0f;
  }
  for (int kt=0; kt<8; ++kt){
    const v16h a = *(const v16h*)(Aswz + ((size_t)(mt*8+kt)*32 + lane)*16);
    #pragma unroll
    for (int t=0;t<16;++t){
      const v16h b = *(const v16h*)(Bg + ((size_t)(kt*128 + (w*16+t))*32 + lane)*16);
      acc[t] = __builtin_amdgcn_wmma_f32_16x16x32_f16(false, a, false, b, (short)0, acc[t], false, false);
    }
  }

  /* ---- LSTM nonlinearity; lane column = (m,g) for s = 16w+t, rows vr + 8*hi */
  #pragma unroll
  for (int t=0;t<16;++t){
    const int s = w*16 + t;
    const float bias = biasP[s*16 + mg];
    #pragma unroll
    for (int vr=0; vr<8; ++vr){
      float val = acc[t][vr] + bias;
      float act = (g==2) ? tanhf(val) : sigf(val);
      int base = (hi<<4) | (m<<2);
      float ai = __shfl(act, base|0, 32);
      float af = __shfl(act, base|1, 32);
      float ag = __shfl(act, base|2, 32);
      float ao = __shfl(act, base|3, 32);
      int row = vr + hi*8;
      float ct  = af*cxL[row*128+s] + ai*ag;
      float htv = ao * tanhf(ct);
      acc[t][vr] = ct;                                   /* keep c_t in regs */
      if (g==0) htL[(row*4+m)*128 + s] = (_Float16)htv;
    }
  }
  __syncthreads();

  /* ---- write_key = h_t @ W_write (per template) */
  for (int rep=0; rep<4; ++rep){
    int o = tid + rep*256;
    int row = o>>6, mm = (o>>4)&3, d = o&15;
    float s = 0.f;
    for (int ss=0; ss<128; ++ss)
      s += (float)htL[(row*4+mm)*128+ss] * W_write[(mm*128+ss)*16 + d];
    wkL[(row*4+mm)*16 + d] = s;
  }
  __syncthreads();

  /* ---- gumbel straight-through selection */
  if (tid < 16){
    int row = tid, n = n0 + row;
    float best = -1e30f; int sel = 0;
    for (int mm=0; mm<4; ++mm){
      float lg = 0.f;
      for (int d=0; d<16; ++d) lg += hreadG[(size_t)n*16+d] * wkL[(row*4+mm)*16+d];
      float y = lg + gumbel_at((uint32_t)(n*4+mm));
      if (y > best){ best = y; sel = mm; }
    }
    selL[row] = sel;
    for (int mm=0; mm<4; ++mm) out[O_TA + (size_t)n*4 + mm] = (mm==sel)?1.0f:0.0f;
  }
  __syncthreads();

  /* ---- selected c_t -> cbL */
  #pragma unroll
  for (int t=0;t<16;++t){
    int s = w*16 + t;
    #pragma unroll
    for (int vr=0; vr<8; ++vr){
      int row = vr + hi*8;
      if (mg == (selL[row]<<2)) cbL[row*128+s] = acc[t][vr];
    }
  }
  __syncthreads();

  /* ---- cx_out */
  for (int r=0;r<8;++r){
    int idx = tid + r*256, row = idx>>7;
    float bm = bmL[row];
    out[O_CXO + (size_t)n0*128 + idx] = bm*cbL[idx] + (1.0f-bm)*cxL[idx];
  }

  /* ---- comm projections (hb @ Wq_c/Wk_c/Wv_c), wave w -> column tile w */
  v8f qa, ka, va;
  #pragma unroll
  for (int i=0;i<8;++i){ qa[i]=0.f; ka[i]=0.f; va[i]=0.f; }
  {
    const int mrow = lane & 15;
    const int sel = selL[mrow];
    for (int kt=0; kt<4; ++kt){
      v16h a;
      #pragma unroll
      for (int i=0;i<8;++i){
        int k = kt*32 + akoff(i,hi);
        a[2*i]   = htL[(mrow*4+sel)*128 + k];
        a[2*i+1] = htL[(mrow*4+sel)*128 + k+1];
      }
      const v16h bq = *(const v16h*)(Wc + (((size_t)(0*4+kt)*8 + w)*32 + lane)*16);
      const v16h bk = *(const v16h*)(Wc + (((size_t)(1*4+kt)*8 + w)*32 + lane)*16);
      const v16h bv = *(const v16h*)(Wc + (((size_t)(2*4+kt)*8 + w)*32 + lane)*16);
      qa = __builtin_amdgcn_wmma_f32_16x16x32_f16(false, a, false, bq, (short)0, qa, false, false);
      ka = __builtin_amdgcn_wmma_f32_16x16x32_f16(false, a, false, bk, (short)0, ka, false, false);
      va = __builtin_amdgcn_wmma_f32_16x16x32_f16(false, a, false, bv, (short)0, va, false, false);
    }
  }
  {
    int col = w*16 + (lane&15);
    #pragma unroll
    for (int vr=0; vr<8; ++vr){
      int row = vr + hi*8;
      qcL[row*128+col] = (_Float16)qa[vr];
      kcL[row*128+col] = (_Float16)ka[vr];
      vcL[row*128+col] = (_Float16)va[vr];
    }
  }
  __syncthreads();

  /* ---- per-batch 4-head attention over 8 units */
  if (tid < 64){
    int bl = tid>>5, rem = tid&31, h = rem>>3, qq = rem&7;
    int rowq = bl*8 + qq;
    float sc[8]; float mx = -1e30f;
    for (int kk=0; kk<8; ++kk){
      int rowk = bl*8 + kk;
      float s = 0.f;
      for (int d=0; d<32; ++d)
        s += (float)qcL[rowq*128 + h*32 + d] * (float)kcL[rowk*128 + h*32 + d];
      s *= 0.17677669529663687f;   /* 1/sqrt(32) */
      sc[kk] = s; mx = fmaxf(mx, s);
    }
    float sum = 0.f;
    for (int kk=0; kk<8; ++kk){ sc[kk] = __expf(sc[kk]-mx); sum += sc[kk]; }
    float inv = 1.0f/sum;
    for (int d=0; d<32; ++d){
      float o = 0.f;
      for (int kk=0; kk<8; ++kk)
        o += sc[kk] * (float)vcL[(bl*8+kk)*128 + h*32 + d];
      oL[rowq*128 + h*32 + d] = o * inv;
    }
  }
  __syncthreads();

  /* ---- gating GEMMs (o @ Wf_c, o @ Wg_c) + final hx_out */
  v8f fa, ga;
  #pragma unroll
  for (int i=0;i<8;++i){ fa[i]=0.f; ga[i]=0.f; }
  {
    const int mrow = lane & 15;
    for (int kt=0; kt<4; ++kt){
      v16h a;
      #pragma unroll
      for (int i=0;i<8;++i){
        int k = kt*32 + akoff(i,hi);
        a[2*i]   = (_Float16)oL[mrow*128 + k];
        a[2*i+1] = (_Float16)oL[mrow*128 + k+1];
      }
      const v16h bfm = *(const v16h*)(Wc + (((size_t)(3*4+kt)*8 + w)*32 + lane)*16);
      const v16h bgm = *(const v16h*)(Wc + (((size_t)(4*4+kt)*8 + w)*32 + lane)*16);
      fa = __builtin_amdgcn_wmma_f32_16x16x32_f16(false, a, false, bfm, (short)0, fa, false, false);
      ga = __builtin_amdgcn_wmma_f32_16x16x32_f16(false, a, false, bgm, (short)0, ga, false, false);
    }
  }
  {
    int col = w*16 + (lane&15);
    float bfv = bf_c[col], bgv = bg_c[col];
    #pragma unroll
    for (int vr=0; vr<8; ++vr){
      int row = vr + hi*8;
      int n = n0 + row;
      float comm = sigf(ga[vr]+bgv) * tanhf(fa[vr]+bfv);
      float hbv = (float)htL[(row*4+selL[row])*128 + col];
      float bm = bmL[row];
      out[O_HX + (size_t)n*128 + col] = bm*(hbv + comm) + (1.0f-bm)*hx[(size_t)n*128 + col];
    }
  }
}

extern "C" void kernel_launch(void* const* d_in, const int* in_sizes, int n_in,
                              void* d_out, int out_size, void* d_ws, size_t ws_size,
                              hipStream_t stream) {
  (void)in_sizes; (void)n_in; (void)out_size; (void)ws_size;
  const float* inp    = (const float*)d_in[0];
  const float* hx     = (const float*)d_in[1];
  const float* cx     = (const float*)d_in[2];
  const float* Wq_i   = (const float*)d_in[3];
  const float* Wk_i   = (const float*)d_in[4];
  const float* Wv_i   = (const float*)d_in[5];
  const float* Wq_c   = (const float*)d_in[6];
  const float* Wk_c   = (const float*)d_in[7];
  const float* Wv_c   = (const float*)d_in[8];
  const float* Wf_c   = (const float*)d_in[9];
  const float* bf_c   = (const float*)d_in[10];
  const float* Wg_c   = (const float*)d_in[11];
  const float* bg_c   = (const float*)d_in[12];
  const float* W_ih   = (const float*)d_in[13];
  const float* W_hh   = (const float*)d_in[14];
  const float* b_ih   = (const float*)d_in[15];
  const float* b_hh   = (const float*)d_in[16];
  const float* W_read = (const float*)d_in[17];
  const float* W_write= (const float*)d_in[18];
  float* out = (float*)d_out;

  char* ws = (char*)d_ws;
  _Float16* Aswz  = (_Float16*)(ws);                                 /* 8 MB  */
  _Float16* Bg    = (_Float16*)(ws + (8u<<20));                      /* 1 MB  */
  _Float16* Wc    = (_Float16*)(ws + (9u<<20));                      /* 320KB */
  float*    biasP = (float*)   (ws + (9u<<20) + (320u<<10));         /* 8 KB  */
  float*    hreadG= (float*)   (ws + (9u<<20) + (328u<<10));         /* 1 MB  */

  k0_swizzle<<<156, 256, 0, stream>>>(W_ih, W_hh, b_ih, b_hh,
                                      Wq_c, Wk_c, Wv_c, Wf_c, Wg_c,
                                      Bg, Wc, biasP);
  k1_inputattn<<<BSZ, 128, 0, stream>>>(inp, hx, Wq_i, Wk_i, Wv_i, W_read,
                                        Aswz, hreadG, out);
  k2_core<<<NTIL, 256, 0, stream>>>(hx, cx, bf_c, bg_c, W_write,
                                    Aswz, Bg, Wc, biasP, hreadG, out);
}
